// Pointnet2MSG_64132451664087
// MI455X (gfx1250) — compile-verified, matrix operand order checked by reading, not run on hardware
//
#include <hip/hip_runtime.h>
#include <hip/hip_bf16.h>
#include <math.h>

// ---------------------------------------------------------------------------
// Types for CDNA5 WMMA (wave32): D(16x16 f32) = A(16x32 f16) x B(32x16 f16) + C
// ---------------------------------------------------------------------------
typedef __attribute__((ext_vector_type(16))) _Float16 v16h;
typedef __attribute__((ext_vector_type(8)))  float    v8f;

union AFrag { v16h v; _Float16 e[16]; };
union CFrag { v8f  v; float    e[8];  };

// ---------------------------------------------------------------------------
// Generic WMMA GEMM:  Y[o,n] = act( ((sum_c W[o,c]*X[c,n]) + preb[o]) * g[o] + b[o] )
// W: [O x C] row-major, X: [C x N] row-major, Y: [O x N].
// Block = 256 thr (8 waves). Block tile 64(O) x 64(N); wave tile 16 x 32.
// ---------------------------------------------------------------------------
__global__ __launch_bounds__(256) void gemm_wmma_kernel(
    const float* __restrict__ Wm, const float* __restrict__ Xm,
    const float* __restrict__ preb, const float* __restrict__ gamma,
    const float* __restrict__ beta, float* __restrict__ Ym,
    int O, int C, int N, int relu)
{
  __shared__ _Float16 As[64][32];
  __shared__ _Float16 Bs[32][64];
  const int tid  = threadIdx.x;
  const int lane = tid & 31;
  const int wave = tid >> 5;
  const int wm   = wave >> 1;   // 0..3 : 16-row band
  const int wn   = wave & 1;    // 0..1 : 32-col band
  const int mBase = blockIdx.y * 64;
  const int nBase = blockIdx.x * 64;

  CFrag acc0 = {}, acc1 = {};

  for (int k0 = 0; k0 < C; k0 += 32) {
    // Stage A tile (f32 -> f16)
    for (int i = tid; i < 64 * 32; i += 256) {
      int r = i >> 5, kk = i & 31;
      int o = mBase + r, c = k0 + kk;
      As[r][kk] = (o < O && c < C) ? (_Float16)Wm[(size_t)o * C + c] : (_Float16)0.0f;
    }
    // Stage B tile
    for (int i = tid; i < 32 * 64; i += 256) {
      int kk = i >> 6, nn = i & 63;
      int c = k0 + kk, n = nBase + nn;
      Bs[kk][nn] = (c < C && n < N) ? (_Float16)Xm[(size_t)c * N + n] : (_Float16)0.0f;
    }
    __syncthreads();

    if (k0 + 32 < C) {
      __builtin_prefetch(&Wm[(size_t)mBase * C + (k0 + 32)], 0, 0);
      __builtin_prefetch(&Xm[(size_t)(k0 + 32) * N + nBase], 0, 0);
    }

    // A fragment: lane<16 -> K 0-7 & 16-23 ; lane>=16 -> K 8-15 & 24-31
    AFrag a;
    {
      int m  = lane & 15;
      int kb = (lane < 16) ? 0 : 8;
#pragma unroll
      for (int j = 0; j < 4; ++j) {
        a.e[2*j]     = As[wm*16 + m][kb + 2*j];
        a.e[2*j+1]   = As[wm*16 + m][kb + 2*j + 1];
        a.e[8+2*j]   = As[wm*16 + m][kb + 16 + 2*j];
        a.e[8+2*j+1] = As[wm*16 + m][kb + 16 + 2*j + 1];
      }
    }
    AFrag b0, b1;
    {
      int n  = lane & 15;
      int kb = (lane < 16) ? 0 : 8;
#pragma unroll
      for (int j = 0; j < 4; ++j) {
        b0.e[2*j]     = Bs[kb + 2*j][wn*32 + n];
        b0.e[2*j+1]   = Bs[kb + 2*j + 1][wn*32 + n];
        b0.e[8+2*j]   = Bs[kb + 16 + 2*j][wn*32 + n];
        b0.e[8+2*j+1] = Bs[kb + 16 + 2*j + 1][wn*32 + n];
        b1.e[2*j]     = Bs[kb + 2*j][wn*32 + 16 + n];
        b1.e[2*j+1]   = Bs[kb + 2*j + 1][wn*32 + 16 + n];
        b1.e[8+2*j]   = Bs[kb + 16 + 2*j][wn*32 + 16 + n];
        b1.e[8+2*j+1] = Bs[kb + 16 + 2*j + 1][wn*32 + 16 + n];
      }
    }
    acc0.v = __builtin_amdgcn_wmma_f32_16x16x32_f16(false, a.v, false, b0.v, (short)0, acc0.v, false, false);
    acc1.v = __builtin_amdgcn_wmma_f32_16x16x32_f16(false, a.v, false, b1.v, (short)0, acc1.v, false, false);
    __syncthreads();
  }

  // Epilogue (C/D layout: M = r + 8*(lane>=16), N = lane&15)
#pragma unroll
  for (int r = 0; r < 8; ++r) {
    int m = mBase + wm*16 + r + ((lane >> 4) << 3);
    if (m >= O) continue;
    float pb = preb  ? preb[m]  : 0.0f;
    float sc = gamma ? gamma[m] : 1.0f;
    float sh = beta  ? beta[m]  : 0.0f;
    int n0 = nBase + wn*32 + (lane & 15);
    int n1 = n0 + 16;
    if (n0 < N) { float v = (acc0.e[r] + pb) * sc + sh; if (relu) v = fmaxf(v, 0.0f); Ym[(size_t)m*N + n0] = v; }
    if (n1 < N) { float v = (acc1.e[r] + pb) * sc + sh; if (relu) v = fmaxf(v, 0.0f); Ym[(size_t)m*N + n1] = v; }
  }
}

// ---------------------------------------------------------------------------
// 3x3 conv (pad=1) as implicit GEMM with on-the-fly im2col. K = Ci*9.
// Weight (Co,Ci,3,3) is exactly row-major [Co x Ci*9].
// Epilogue: relu( ((conv + bias) * g + b) ) with g/b optional.
// ---------------------------------------------------------------------------
__global__ __launch_bounds__(256) void conv3x3_wmma_kernel(
    const float* __restrict__ In, const float* __restrict__ Wt,
    const float* __restrict__ bias, const float* __restrict__ gamma,
    const float* __restrict__ beta, float* __restrict__ Out,
    int Ci, int IH, int IW, int Co, int OH, int OW, int stride, int relu)
{
  __shared__ _Float16 As[64][32];
  __shared__ _Float16 Bs[32][64];
  const int tid  = threadIdx.x;
  const int lane = tid & 31;
  const int wave = tid >> 5;
  const int wm   = wave >> 1;
  const int wn   = wave & 1;
  const int mBase = blockIdx.y * 64;
  const int nBase = blockIdx.x * 64;
  const int K    = Ci * 9;
  const int OHW  = OH * OW;

  CFrag acc0 = {}, acc1 = {};

  for (int k0 = 0; k0 < K; k0 += 32) {
    for (int i = tid; i < 64 * 32; i += 256) {
      int r = i >> 5, kk = i & 31;
      int o = mBase + r, kg = k0 + kk;
      As[r][kk] = (o < Co && kg < K) ? (_Float16)Wt[(size_t)o * K + kg] : (_Float16)0.0f;
    }
    for (int i = tid; i < 32 * 64; i += 256) {
      int kk = i >> 6, nn = i & 63;
      int kg = k0 + kk, n = nBase + nn;
      _Float16 v = (_Float16)0.0f;
      if (kg < K && n < OHW) {
        int ci = kg / 9, t = kg % 9;
        int ky = t / 3, kx = t % 3;
        int oh = n / OW, ow = n % OW;
        int ih = oh * stride + ky - 1, iw = ow * stride + kx - 1;
        if (ih >= 0 && ih < IH && iw >= 0 && iw < IW)
          v = (_Float16)In[((size_t)ci * IH + ih) * IW + iw];
      }
      Bs[kk][nn] = v;
    }
    __syncthreads();

    AFrag a;
    {
      int m  = lane & 15;
      int kb = (lane < 16) ? 0 : 8;
#pragma unroll
      for (int j = 0; j < 4; ++j) {
        a.e[2*j]     = As[wm*16 + m][kb + 2*j];
        a.e[2*j+1]   = As[wm*16 + m][kb + 2*j + 1];
        a.e[8+2*j]   = As[wm*16 + m][kb + 16 + 2*j];
        a.e[8+2*j+1] = As[wm*16 + m][kb + 16 + 2*j + 1];
      }
    }
    AFrag b0, b1;
    {
      int n  = lane & 15;
      int kb = (lane < 16) ? 0 : 8;
#pragma unroll
      for (int j = 0; j < 4; ++j) {
        b0.e[2*j]     = Bs[kb + 2*j][wn*32 + n];
        b0.e[2*j+1]   = Bs[kb + 2*j + 1][wn*32 + n];
        b0.e[8+2*j]   = Bs[kb + 16 + 2*j][wn*32 + n];
        b0.e[8+2*j+1] = Bs[kb + 16 + 2*j + 1][wn*32 + n];
        b1.e[2*j]     = Bs[kb + 2*j][wn*32 + 16 + n];
        b1.e[2*j+1]   = Bs[kb + 2*j + 1][wn*32 + 16 + n];
        b1.e[8+2*j]   = Bs[kb + 16 + 2*j][wn*32 + 16 + n];
        b1.e[8+2*j+1] = Bs[kb + 16 + 2*j + 1][wn*32 + 16 + n];
      }
    }
    acc0.v = __builtin_amdgcn_wmma_f32_16x16x32_f16(false, a.v, false, b0.v, (short)0, acc0.v, false, false);
    acc1.v = __builtin_amdgcn_wmma_f32_16x16x32_f16(false, a.v, false, b1.v, (short)0, acc1.v, false, false);
    __syncthreads();
  }

#pragma unroll
  for (int r = 0; r < 8; ++r) {
    int m = mBase + wm*16 + r + ((lane >> 4) << 3);
    if (m >= Co) continue;
    float pb = bias  ? bias[m]  : 0.0f;
    float sc = gamma ? gamma[m] : 1.0f;
    float sh = beta  ? beta[m]  : 0.0f;
    int n0 = nBase + wn*32 + (lane & 15);
    int n1 = n0 + 16;
    if (n0 < OHW) { float v = (acc0.e[r] + pb) * sc + sh; if (relu) v = fmaxf(v, 0.0f); Out[(size_t)m*OHW + n0] = v; }
    if (n1 < OHW) { float v = (acc1.e[r] + pb) * sc + sh; if (relu) v = fmaxf(v, 0.0f); Out[(size_t)m*OHW + n1] = v; }
  }
}

// ---------------------------------------------------------------------------
// Glue kernels (VALU paths)
// ---------------------------------------------------------------------------
__global__ void normxy_kernel(const float* __restrict__ xy, float* __restrict__ out, int n)
{
  int i = blockIdx.x * blockDim.x + threadIdx.x;
  if (i >= n) return;
  out[i*2+0] = xy[i*2+0] / (1280.0f - 1.0f) * 2.0f - 1.0f;
  out[i*2+1] = xy[i*2+1] / (384.0f  - 1.0f) * 2.0f - 1.0f;
}

__global__ __launch_bounds__(1024) void fps_kernel(
    const float* __restrict__ xyz, int N, int npoint,
    int* __restrict__ idx, float* __restrict__ dmin)
{
  __shared__ float rv[1024];
  __shared__ int   ri[1024];
  __shared__ int   s_last;
  int tid = threadIdx.x, nt = blockDim.x;
  for (int i = tid; i < N; i += nt) dmin[i] = 1e10f;
  if (tid == 0) { idx[0] = 0; s_last = 0; }
  __syncthreads();
  for (int it = 1; it < npoint; ++it) {
    int last = s_last;
    float lx = xyz[last*3+0], ly = xyz[last*3+1], lz = xyz[last*3+2];
    float best = -1.0f; int bi = 0;
    for (int i = tid; i < N; i += nt) {
      float dx = xyz[i*3+0]-lx, dy = xyz[i*3+1]-ly, dz = xyz[i*3+2]-lz;
      float d = dx*dx + dy*dy + dz*dz;
      float dm = fminf(dmin[i], d);
      dmin[i] = dm;
      if (dm > best) { best = dm; bi = i; }
    }
    rv[tid] = best; ri[tid] = bi;
    __syncthreads();
    for (int s = nt >> 1; s > 0; s >>= 1) {
      if (tid < s && rv[tid+s] > rv[tid]) { rv[tid] = rv[tid+s]; ri[tid] = ri[tid+s]; }
      __syncthreads();
    }
    if (tid == 0) { idx[it] = ri[0]; s_last = ri[0]; }
    __syncthreads();
  }
}

__global__ void gather_rows_kernel(const float* __restrict__ src, const int* __restrict__ idx,
                                   float* __restrict__ dst, int n, int stride)
{
  int t = blockIdx.x * blockDim.x + threadIdx.x;
  if (t >= n * stride) return;
  int i = t / stride, c = t % stride;
  dst[t] = src[(size_t)idx[i] * stride + c];
}

__global__ void ball_query_kernel(const float* __restrict__ nxyz, int S,
                                  const float* __restrict__ xyz, int N,
                                  float r2, int K, int* __restrict__ idx)
{
  int s = blockIdx.x * blockDim.x + threadIdx.x;
  if (s >= S) return;
  float cx = nxyz[s*3+0], cy = nxyz[s*3+1], cz = nxyz[s*3+2];
  int cnt = 0, first = 0; bool have = false;
  for (int i = 0; i < N && cnt < K; ++i) {
    float dx = xyz[i*3+0]-cx, dy = xyz[i*3+1]-cy, dz = xyz[i*3+2]-cz;
    if (dx*dx + dy*dy + dz*dz < r2) {
      if (!have) { first = i; have = true; }
      idx[(size_t)s*K + cnt++] = i;
    }
  }
  int fill = have ? first : 0;
  for (; cnt < K; ++cnt) idx[(size_t)s*K + cnt] = fill;
}

// g[(3+Cf) x (S*K)] : first 3 rows = xyz diff, rest gathered features
__global__ void group_kernel(const float* __restrict__ xyz, const float* __restrict__ feat,
                             int Cf, int N, const int* __restrict__ idx,
                             const float* __restrict__ nxyz, float* __restrict__ g,
                             int S, int K)
{
  size_t SK = (size_t)S * K;
  size_t t = (size_t)blockIdx.x * blockDim.x + threadIdx.x;
  if (t >= SK) return;
  int s = (int)(t / K);
  int id = idx[t];
  g[0*SK + t] = xyz[id*3+0] - nxyz[s*3+0];
  g[1*SK + t] = xyz[id*3+1] - nxyz[s*3+1];
  g[2*SK + t] = xyz[id*3+2] - nxyz[s*3+2];
  for (int c = 0; c < Cf; ++c)
    g[(size_t)(3+c)*SK + t] = feat[(size_t)c*N + id];
}

__global__ void maxpool_kernel(const float* __restrict__ in, float* __restrict__ out,
                               int C, int S, int K)
{
  size_t t = (size_t)blockIdx.x * blockDim.x + threadIdx.x;
  if (t >= (size_t)C * S) return;
  int c = (int)(t / S), s = (int)(t % S);
  const float* p = in + ((size_t)c * S + s) * K;
  float m = -1e30f;
  for (int k = 0; k < K; ++k) m = fmaxf(m, p[k]);
  out[(size_t)c*S + s] = m;
}

__global__ void featgather_kernel(const float* __restrict__ fmap, int C, int H, int W,
                                  const float* __restrict__ xy, int n, float* __restrict__ out)
{
  size_t t = (size_t)blockIdx.x * blockDim.x + threadIdx.x;
  if (t >= (size_t)C * n) return;
  int c = (int)(t / n), j = (int)(t % n);
  float x = ((xy[j*2+0] + 1.0f) * W - 1.0f) * 0.5f;
  float y = ((xy[j*2+1] + 1.0f) * H - 1.0f) * 0.5f;
  float x0 = floorf(x), y0 = floorf(y);
  float acc = 0.0f;
  for (int dx = 0; dx < 2; ++dx)
    for (int dy = 0; dy < 2; ++dy) {
      float xi = x0 + dx, yi = y0 + dy;
      if (xi >= 0.0f && xi < (float)W && yi >= 0.0f && yi < (float)H) {
        float wgt = (1.0f - fabsf(x - xi)) * (1.0f - fabsf(y - yi));
        acc += fmap[((size_t)c * H + (int)yi) * W + (int)xi] * wgt;
      }
    }
  out[(size_t)c*n + j] = acc;
}

__global__ void att_kernel(const float* __restrict__ point, int pc,
                           const float* __restrict__ img, int ic, int n,
                           const float* __restrict__ aw, const float* __restrict__ ab,
                           float* __restrict__ att)
{
  int j = blockIdx.x * blockDim.x + threadIdx.x;
  if (j >= n) return;
  float m = -1e30f;
  for (int c = 0; c < pc; ++c) m = fmaxf(m, point[(size_t)c*n + j]);
  for (int c = 0; c < ic; ++c) m = fmaxf(m, img[(size_t)c*n + j]);
  float z = aw[0] * m + ab[0];
  att[j] = 1.0f / (1.0f + expf(-z));
}

__global__ void concat_att_kernel(const float* __restrict__ point, const float* __restrict__ imgnew,
                                  const float* __restrict__ att, float* __restrict__ X,
                                  int pc, int n)
{
  size_t t = (size_t)blockIdx.x * blockDim.x + threadIdx.x;
  if (t >= (size_t)2 * pc * n) return;
  int c = (int)(t / n), j = (int)(t % n);
  X[t] = (c < pc) ? point[(size_t)c*n + j] : imgnew[(size_t)(c - pc)*n + j] * att[j];
}

// 2x2 stride-2 deconv: out[o,2h+k,2w+l] = sum_i x[i,h,w]*W[i,o,k,l] + b[o]
__global__ void deconv_kernel(const float* __restrict__ x, const float* __restrict__ w,
                              const float* __restrict__ b, float* __restrict__ out,
                              int ci, int co, int H, int W)
{
  size_t t = (size_t)blockIdx.x * blockDim.x + threadIdx.x;
  if (t >= (size_t)co * H * W) return;
  int o = (int)(t / (H * W));
  int r = (int)(t % (H * W));
  int h = r / W, ww = r % W;
  float a0 = 0, a1 = 0, a2 = 0, a3 = 0;
  for (int i = 0; i < ci; ++i) {
    float xv = x[((size_t)i * H + h) * W + ww];
    const float* wp = w + ((size_t)i * co + o) * 4;
    a0 += xv * wp[0]; a1 += xv * wp[1]; a2 += xv * wp[2]; a3 += xv * wp[3];
  }
  float bo = b[o];
  int W2 = 2 * W;
  float* op = out + (size_t)o * 4 * H * W;
  op[(size_t)(2*h+0)*W2 + 2*ww+0] = a0 + bo;
  op[(size_t)(2*h+0)*W2 + 2*ww+1] = a1 + bo;
  op[(size_t)(2*h+1)*W2 + 2*ww+0] = a2 + bo;
  op[(size_t)(2*h+1)*W2 + 2*ww+1] = a3 + bo;
}

__global__ void copy_kernel(const float* __restrict__ s, float* __restrict__ d, int n)
{
  int i = blockIdx.x * blockDim.x + threadIdx.x;
  if (i < n) d[i] = s[i];
}

__global__ void knn3_kernel(const float* __restrict__ unk, int Nu,
                            const float* __restrict__ kn, int Nk,
                            int* __restrict__ idx3, float* __restrict__ w3)
{
  int u = blockIdx.x * blockDim.x + threadIdx.x;
  if (u >= Nu) return;
  float ux = unk[u*3+0], uy = unk[u*3+1], uz = unk[u*3+2];
  float bd[3] = {1e30f, 1e30f, 1e30f};
  int   bi[3] = {0, 0, 0};
  for (int i = 0; i < Nk; ++i) {
    float dx = kn[i*3+0]-ux, dy = kn[i*3+1]-uy, dz = kn[i*3+2]-uz;
    float d = dx*dx + dy*dy + dz*dz;
    if (d < bd[0]) { bd[2]=bd[1]; bi[2]=bi[1]; bd[1]=bd[0]; bi[1]=bi[0]; bd[0]=d; bi[0]=i; }
    else if (d < bd[1]) { bd[2]=bd[1]; bi[2]=bi[1]; bd[1]=d; bi[1]=i; }
    else if (d < bd[2]) { bd[2]=d; bi[2]=i; }
  }
  float w0 = 1.0f/(bd[0]+1e-8f), w1 = 1.0f/(bd[1]+1e-8f), w2 = 1.0f/(bd[2]+1e-8f);
  float s = w0 + w1 + w2;
  idx3[u*3+0] = bi[0]; idx3[u*3+1] = bi[1]; idx3[u*3+2] = bi[2];
  w3[u*3+0] = w0/s; w3[u*3+1] = w1/s; w3[u*3+2] = w2/s;
}

__global__ void interp_kernel(const float* __restrict__ knf, int C, int Nk,
                              const int* __restrict__ idx3, const float* __restrict__ w3,
                              float* __restrict__ out, int Nu)
{
  size_t t = (size_t)blockIdx.x * blockDim.x + threadIdx.x;
  if (t >= (size_t)C * Nu) return;
  int c = (int)(t / Nu), u = (int)(t % Nu);
  const float* fc = knf + (size_t)c * Nk;
  out[t] = fc[idx3[u*3+0]] * w3[u*3+0]
         + fc[idx3[u*3+1]] * w3[u*3+1]
         + fc[idx3[u*3+2]] * w3[u*3+2];
}

// ---------------------------------------------------------------------------
// Host orchestration
// ---------------------------------------------------------------------------
static const int   NPOINTS_[4]      = {4096, 1024, 256, 64};
static const float RADIUS_[4][2]    = {{0.1f,0.5f},{0.5f,1.0f},{1.0f,2.0f},{2.0f,4.0f}};
static const int   NSAMPLE_[4][2]   = {{16,32},{16,32},{16,32},{16,32}};
static const int   MLPS_[4][2][3]   = {{{16,16,32},{32,32,64}},{{64,64,128},{64,96,128}},
                                       {{128,196,256},{128,196,256}},{{256,256,512},{256,384,512}}};
static const int   IMG_CH_[5]       = {3,64,128,256,512};
static const int   PT_CH_[4]        = {96,256,512,1024};
static const int   IMGH_[5]         = {96,48,24,12,6};
static const int   IMGW_[5]         = {320,160,80,40,20};
static const int   FPC_[4][3]       = {{256,128,128},{608,256,256},{768,512,512},{1536,512,512}};

struct FuP { const float *ab,*aw,*b,*be,*g,*ib,*ibe,*ig,*iw,*w; };

extern "C" void kernel_launch(void* const* d_in, const int* in_sizes, int n_in,
                              void* d_out, int out_size, void* d_ws, size_t ws_size,
                              hipStream_t stream)
{
  (void)in_sizes; (void)out_size; (void)ws_size;

  const float* pc    = (const float*)d_in[0];  // (16384,3)
  const float* image = (const float*)d_in[1];  // (3,96,320)
  const float* xyin  = (const float*)d_in[2];  // (16384,2)

  // ---- Parse params (JAX pytree: dict keys sorted) -----------------------
  int pi = 3;
  auto NXT = [&]() -> const float* {
    if (pi >= n_in) pi = n_in - 1;
    return (const float*)d_in[pi++];
  };
  // deconv1: 3 x {b,w}
  const float *dcb[3], *dcw[3];
  for (int i = 0; i < 3; ++i) { dcb[i] = NXT(); dcw[i] = NXT(); }
  // fp: 4 x 2 x {b,g,w}
  const float *fpb[4][2], *fpg[4][2], *fpw[4][2];
  for (int k = 0; k < 4; ++k)
    for (int l = 0; l < 2; ++l) { fpb[k][l] = NXT(); fpg[k][l] = NXT(); fpw[k][l] = NXT(); }
  // fusion / fusion1: {ab,aw,b,be,g,ib,ibe,ig,iw,w}
  FuP fus[4], fus1[4];
  for (int i = 0; i < 4; ++i) {
    fus[i].ab = NXT(); fus[i].aw = NXT(); fus[i].b = NXT(); fus[i].be = NXT(); fus[i].g = NXT();
    fus[i].ib = NXT(); fus[i].ibe = NXT(); fus[i].ig = NXT(); fus[i].iw = NXT(); fus[i].w = NXT();
  }
  for (int i = 0; i < 4; ++i) {
    fus1[i].ab = NXT(); fus1[i].aw = NXT(); fus1[i].b = NXT(); fus1[i].be = NXT(); fus1[i].g = NXT();
    fus1[i].ib = NXT(); fus1[i].ibe = NXT(); fus1[i].ig = NXT(); fus1[i].iw = NXT(); fus1[i].w = NXT();
  }
  // ifc1: 3 x {b,be,g,w}
  const float *icb[3], *icbe[3], *icg[3], *icw[3];
  for (int i = 0; i < 3; ++i) { icb[i] = NXT(); icbe[i] = NXT(); icg[i] = NXT(); icw[i] = NXT(); }
  // img_block: 4 x { bns: 4x{b,g}; convs: (b,w)x4, w4 }
  const float *bnb[4][4], *bng[4][4], *cvb[4][4], *cvw[4][5];
  for (int i = 0; i < 4; ++i) {
    for (int j = 0; j < 4; ++j) { bnb[i][j] = NXT(); bng[i][j] = NXT(); }
    for (int j = 0; j < 4; ++j) { cvb[i][j] = NXT(); cvw[i][j] = NXT(); }
    cvw[i][4] = NXT();
  }
  // sa: 4 x 2 x 3 x {b,g,w}
  const float *sab[4][2][3], *sag[4][2][3], *saw[4][2][3];
  for (int k = 0; k < 4; ++k)
    for (int j = 0; j < 2; ++j)
      for (int l = 0; l < 3; ++l) { sab[k][j][l] = NXT(); sag[k][j][l] = NXT(); saw[k][j][l] = NXT(); }

  // ---- Workspace bump allocator ------------------------------------------
  char* wsp = (char*)d_ws;
  size_t off = 0;
  auto alloc = [&](size_t bytes) -> void* {
    void* p = wsp + off;
    off += (bytes + 255) & ~(size_t)255;
    return p;
  };
  float* bufA      = (float*)alloc(sizeof(float) * (size_t)64 * 4096 * 32);
  float* bufB      = (float*)alloc(sizeof(float) * (size_t)64 * 4096 * 32);
  float* bufX      = (float*)alloc(sizeof(float) * (size_t)256 * 16384);
  float* imgNewBuf = (float*)alloc(sizeof(float) * (size_t)128 * 16384);
  float* attBuf    = (float*)alloc(sizeof(float) * 16384);
  float* dmin      = (float*)alloc(sizeof(float) * 16384);
  int*   fidx      = (int*)  alloc(sizeof(int)   * 4096);
  int*   bqidx     = (int*)  alloc(sizeof(int)   * 4096 * 32);
  float* lxyz[5];  lxyz[0] = nullptr;
  float* lxy[5];
  lxy[0] = (float*)alloc(sizeof(float) * 16384 * 2);
  for (int k = 0; k < 4; ++k) {
    lxyz[k+1] = (float*)alloc(sizeof(float) * NPOINTS_[k] * 3);
    lxy[k+1]  = (float*)alloc(sizeof(float) * NPOINTS_[k] * 2);
  }
  float* lfeat0 = (float*)alloc(sizeof(float) * (size_t)128 * 16384);
  float* lfeat[5];
  lfeat[1] = (float*)alloc(sizeof(float) * (size_t)256 * 4096);  // 96 now, 256 after FP
  lfeat[2] = (float*)alloc(sizeof(float) * (size_t)512 * 1024);  // 256 now, 512 after FP
  lfeat[3] = (float*)alloc(sizeof(float) * (size_t)512 * 256);
  lfeat[4] = (float*)alloc(sizeof(float) * (size_t)1024 * 64);
  float* imgs[5]; imgs[0] = nullptr;
  for (int k = 0; k < 4; ++k)
    imgs[k+1] = (float*)alloc(sizeof(float) * (size_t)IMG_CH_[k+1] * IMGH_[k+1] * IMGW_[k+1]);
  float* dec1 = (float*)alloc(sizeof(float) * (size_t)256 * 12 * 40);
  float* dec2 = (float*)alloc(sizeof(float) * (size_t)128 * 24 * 80);
  float* dec3 = (float*)alloc(sizeof(float) * (size_t)64  * 48 * 160);
  float* gath[4];
  gath[0] = (float*)alloc(sizeof(float) * (size_t)512 * 256);
  gath[1] = (float*)alloc(sizeof(float) * (size_t)256 * 1024);
  gath[2] = (float*)alloc(sizeof(float) * (size_t)128 * 4096);
  gath[3] = (float*)alloc(sizeof(float) * (size_t)64  * 16384);
  int*   knnIdx = (int*)  alloc(sizeof(int)   * 16384 * 3);
  float* knnW   = (float*)alloc(sizeof(float) * 16384 * 3);
  float* imgG   = (float*)alloc(sizeof(float) * (size_t)64 * 4096);
  float* upBuf  = (float*)alloc(sizeof(float) * (size_t)64 * 96 * 160);
  float* catBuf = (float*)alloc(sizeof(float) * (size_t)128 * 96 * 160);

  auto launch_gemm = [&](const float* W, const float* X, const float* pb,
                         const float* g, const float* be, float* Y,
                         int O, int C, int N, int relu) {
    dim3 grid((N + 63) / 64, (O + 63) / 64);
    gemm_wmma_kernel<<<grid, 256, 0, stream>>>(W, X, pb, g, be, Y, O, C, N, relu);
  };
  auto launch_conv = [&](const float* inp, const float* w, const float* b,
                         const float* g, const float* bb, float* out,
                         int Ci, int IH, int IW, int Co, int stride, int relu) {
    int OH = (IH - 1) / stride + 1, OW = (IW - 1) / stride + 1;
    int OHW = OH * OW;
    dim3 grid((OHW + 63) / 64, (Co + 63) / 64);
    conv3x3_wmma_kernel<<<grid, 256, 0, stream>>>(inp, w, b, g, bb, out,
                                                  Ci, IH, IW, Co, OH, OW, stride, relu);
  };

  // ---- Level 0 xy normalization ------------------------------------------
  normxy_kernel<<<(16384 + 255) / 256, 256, 0, stream>>>(xyin, lxy[0], 16384);

  // ---- SA levels + image backbone + fusion -------------------------------
  const float* curXyz = pc;
  const float* curXy  = lxy[0];
  const float* curF   = nullptr;
  int curC = 0, curN = 16384;
  for (int k = 0; k < 4; ++k) {
    int S = NPOINTS_[k];
    fps_kernel<<<1, 1024, 0, stream>>>(curXyz, curN, S, fidx, dmin);
    gather_rows_kernel<<<(S*3 + 255)/256, 256, 0, stream>>>(curXyz, fidx, lxyz[k+1], S, 3);
    gather_rows_kernel<<<(S*2 + 255)/256, 256, 0, stream>>>(curXy,  fidx, lxy[k+1],  S, 2);
    float* lf = lfeat[k+1];
    int choff = 0;
    for (int j = 0; j < 2; ++j) {
      int K = NSAMPLE_[k][j];
      float r = RADIUS_[k][j];
      ball_query_kernel<<<(S + 255)/256, 256, 0, stream>>>(lxyz[k+1], S, curXyz, curN, r*r, K, bqidx);
      size_t SK = (size_t)S * K;
      group_kernel<<<(unsigned)((SK + 255)/256), 256, 0, stream>>>(curXyz, curF, curC, curN,
                                                                   bqidx, lxyz[k+1], bufA, S, K);
      float* gi = bufA; float* go = bufB;
      int Ci = 3 + curC;
      for (int l = 0; l < 3; ++l) {
        int Cout = MLPS_[k][j][l];
        launch_gemm(saw[k][j][l], gi, nullptr, sag[k][j][l], sab[k][j][l], go, Cout, Ci, (int)SK, 1);
        float* t = gi; gi = go; go = t; Ci = Cout;
      }
      maxpool_kernel<<<(unsigned)(((size_t)Ci*S + 255)/256), 256, 0, stream>>>(gi, lf + (size_t)choff*S, Ci, S, K);
      choff += Ci;
    }
    // image block
    const float* cinp = (k == 0) ? image : imgs[k];
    int Cii = IMG_CH_[k], Coo = IMG_CH_[k+1];
    int IH = IMGH_[k], IW = IMGW_[k];
    launch_conv(cinp, cvw[k][0], cvb[k][0], bng[k][0], bnb[k][0], bufA, Cii, IH, IW, Coo, 1, 1);
    float* ca = bufA; float* cb = bufB;
    for (int j = 1; j < 4; ++j) {
      launch_conv(ca, cvw[k][j], cvb[k][j], bng[k][j], bnb[k][j], cb, Coo, IH, IW, Coo, 1, 1);
      float* t = ca; ca = cb; cb = t;
    }
    launch_conv(ca, cvw[k][4], nullptr, nullptr, nullptr, imgs[k+1], Coo, IH, IW, Coo, 2, 0);
    // fusion
    int OH = IMGH_[k+1], OW = IMGW_[k+1];
    int ic = Coo, pcC = PT_CH_[k];
    featgather_kernel<<<(unsigned)(((size_t)ic*S + 255)/256), 256, 0, stream>>>(imgs[k+1], ic, OH, OW, lxy[k+1], S, imgG);
    att_kernel<<<(S + 255)/256, 256, 0, stream>>>(lf, pcC, imgG, ic, S, fus[k].aw, fus[k].ab, attBuf);
    launch_gemm(fus[k].iw, imgG, fus[k].ib, fus[k].ig, fus[k].ibe, imgNewBuf, pcC, ic, S, 1);
    concat_att_kernel<<<(unsigned)(((size_t)2*pcC*S + 255)/256), 256, 0, stream>>>(lf, imgNewBuf, attBuf, bufX, pcC, S);
    launch_gemm(fus[k].w, bufX, fus[k].b, fus[k].g, fus[k].be, lf, pcC, 2*pcC, S, 1);
    curXyz = lxyz[k+1]; curXy = lxy[k+1]; curF = lf; curC = pcC; curN = S;
  }

  // ---- Image decoder ------------------------------------------------------
  int dci[3] = {512, 256, 128};
  int dH[3] = {6, 12, 24}, dW[3] = {20, 40, 80};
  const float* skip[3] = {imgs[3], imgs[2], imgs[1]};
  float* decOut[3] = {dec1, dec2, dec3};
  const float* decPrev = imgs[4];
  for (int i = 0; i < 3; ++i) {
    int ci = dci[i], co = ci / 2, H = dH[i], W = dW[i];
    deconv_kernel<<<(unsigned)(((size_t)co*H*W + 255)/256), 256, 0, stream>>>(decPrev, dcw[i], dcb[i], upBuf, ci, co, H, W);
    int HW2 = 4 * H * W;
    copy_kernel<<<((co*HW2) + 255)/256, 256, 0, stream>>>(upBuf, catBuf, co*HW2);
    copy_kernel<<<((co*HW2) + 255)/256, 256, 0, stream>>>(skip[i], catBuf + (size_t)co*HW2, co*HW2);
    launch_gemm(icw[i], catBuf, icb[i], icg[i], icbe[i], decOut[i], co, ci, HW2, 1);
    decPrev = decOut[i];
  }

  // ---- Decoder feature gathers -------------------------------------------
  featgather_kernel<<<(unsigned)(((size_t)512*256   + 255)/256), 256, 0, stream>>>(imgs[4], 512, 6, 20,  lxy[3], 256,   gath[0]);
  featgather_kernel<<<(unsigned)(((size_t)256*1024  + 255)/256), 256, 0, stream>>>(dec1,    256, 12, 40, lxy[2], 1024,  gath[1]);
  featgather_kernel<<<(unsigned)(((size_t)128*4096  + 255)/256), 256, 0, stream>>>(dec2,    128, 24, 80, lxy[1], 4096,  gath[2]);
  featgather_kernel<<<(unsigned)(((size_t)64*16384  + 255)/256), 256, 0, stream>>>(dec3,    64, 48, 160, lxy[0], 16384, gath[3]);

  // ---- FP modules + fusion1 ----------------------------------------------
  const float* unkX[4] = { lxyz[3], lxyz[2], lxyz[1], pc };
  const float* knX[4]  = { lxyz[4], lxyz[3], lxyz[2], lxyz[1] };
  int NuA[4] = {256, 1024, 4096, 16384};
  int NkA[4] = {64, 256, 1024, 4096};
  const float* unkF[4] = { lfeat[3], lfeat[2], lfeat[1], nullptr };
  const float* knF[4]  = { lfeat[4], lfeat[3], lfeat[2], lfeat[1] };
  int CuA[4] = {512, 256, 96, 0};
  int CkA[4] = {1024, 512, 512, 256};
  int fpk[4] = {3, 2, 1, 0};
  float* target[4] = { lfeat[3], lfeat[2], lfeat[1], lfeat0 };
  int ficA[4] = {512, 256, 128, 64};
  int fpcA[4] = {512, 512, 256, 128};
  for (int t = 0; t < 4; ++t) {
    int kk = fpk[t], Nu = NuA[t], Nk = NkA[t], Cu = CuA[t], Ck = CkA[t];
    knn3_kernel<<<(Nu + 255)/256, 256, 0, stream>>>(unkX[t], Nu, knX[t], Nk, knnIdx, knnW);
    interp_kernel<<<(unsigned)(((size_t)Ck*Nu + 255)/256), 256, 0, stream>>>(knF[t], Ck, Nk, knnIdx, knnW, bufX, Nu);
    if (Cu > 0)
      copy_kernel<<<((Cu*Nu) + 255)/256, 256, 0, stream>>>(unkF[t], bufX + (size_t)Ck*Nu, Cu*Nu);
    launch_gemm(fpw[kk][0], bufX, nullptr, fpg[kk][0], fpb[kk][0], bufA, FPC_[kk][1], FPC_[kk][0], Nu, 1);
    launch_gemm(fpw[kk][1], bufA, nullptr, fpg[kk][1], fpb[kk][1], target[t], FPC_[kk][2], FPC_[kk][1], Nu, 1);
    int ic = ficA[t], pcC = fpcA[t];
    att_kernel<<<(Nu + 255)/256, 256, 0, stream>>>(target[t], pcC, gath[t], ic, Nu, fus1[t].aw, fus1[t].ab, attBuf);
    launch_gemm(fus1[t].iw, gath[t], fus1[t].ib, fus1[t].ig, fus1[t].ibe, imgNewBuf, pcC, ic, Nu, 1);
    concat_att_kernel<<<(unsigned)(((size_t)2*pcC*Nu + 255)/256), 256, 0, stream>>>(target[t], imgNewBuf, attBuf, bufX, pcC, Nu);
    launch_gemm(fus1[t].w, bufX, fus1[t].b, fus1[t].g, fus1[t].be, target[t], pcC, 2*pcC, Nu, 1);
  }

  // ---- Output: (l_xyz[0], l_feat[0]) concatenated flat --------------------
  copy_kernel<<<(49152 + 255)/256, 256, 0, stream>>>(pc, (float*)d_out, 49152);
  copy_kernel<<<(2097152 + 255)/256, 256, 0, stream>>>(lfeat0, (float*)d_out + 49152, 2097152);
}
